// DualProcessThinkingFinal_55697135895182
// MI455X (gfx1250) — matrix operand order repaired and checked by reading, not compiled, not run on hardware
//
#include <hip/hip_runtime.h>
#include <hip/hip_bf16.h>

// ---------------------------------------------------------------------------
// Dual-process transformer for MI455X (gfx1250), wave32 + WMMA f32_16x16x32_f16
// + async global->LDS copies (ASYNCcnt) for attention Q/K tiles.
// ---------------------------------------------------------------------------

typedef __attribute__((ext_vector_type(16))) _Float16 h16v;
typedef __attribute__((ext_vector_type(8)))  _Float16 h8v;
typedef __attribute__((ext_vector_type(8)))  float    f8v;

#define HTOT  512
#define SEQ   512
#define BATCH 16
#define NROWS (BATCH * SEQ)   // 8192 token rows

__device__ inline h16v mk16(h8v lo, h8v hi) {
  h16v r;
#pragma unroll
  for (int i = 0; i < 8; ++i) { r[i] = lo[i]; r[i + 8] = hi[i]; }
  return r;
}

__device__ inline f8v wmma32(h16v a, h16v b, f8v c) {
  return __builtin_amdgcn_wmma_f32_16x16x32_f16(false, a, false, b, (short)0, c,
                                                false, false);
}

__device__ inline float gelu_tanh(float x) {
  const float k0 = 0.7978845608028654f;
  float u = k0 * (x + 0.044715f * x * x * x);
  return 0.5f * x * (1.0f + tanhf(u));
}

// Async DMA of 16 bytes global -> LDS (per lane), tracked by ASYNCcnt.
// LDS byte offset = low 32 bits of the generic shared-aperture address
// (ISA 10.2).
__device__ inline void async_copy_b128(void* lds_dst, const void* gaddr) {
  uint32_t l = (uint32_t)(uintptr_t)lds_dst;
  asm volatile("global_load_async_to_lds_b128 %0, %1, off"
               :: "v"(l), "v"((uint64_t)(uintptr_t)gaddr)
               : "memory");
}
__device__ inline void wait_async0() {
  asm volatile("s_wait_asynccnt 0" ::: "memory");
}

// ---------------------------------------------------------------------------
// GEMM: C[M,N] = A[M,K](f16 row-major) * Bt[N,K](f16 row-major, i.e. B^T) + bias
// MODE 1: f32 out = acc + bias + res     (residual fuse)
// MODE 2: f16 out = acc + bias           (q/k/v projections)
// MODE 3: f16 out = gelu(acc + bias)     (MLP up-projection)
// Block tile 128x128, 8 waves (4M x 2N), each wave 32x64 = 2x4 WMMA tiles.
// ---------------------------------------------------------------------------
#define BM  128
#define BN  128
#define BKK 32
#define LDT 40   // BKK + 8 halves pad (80B rows, 16B aligned)

template<int MODE>
__global__ __launch_bounds__(256) void k_gemm(
    const _Float16* __restrict__ A, const _Float16* __restrict__ Bt,
    const float* __restrict__ bias, const float* __restrict__ res,
    float* __restrict__ outF, _Float16* __restrict__ outH, int N, int K)
{
  __shared__ _Float16 As[2][BM * LDT];
  __shared__ _Float16 Bs[2][BN * LDT];
  const int tid  = threadIdx.x;
  const int lane = tid & 31;
  const int wave = tid >> 5;
  const int wm   = wave >> 1;      // 0..3
  const int wn   = wave & 1;       // 0..1
  const int m0   = blockIdx.y * BM;
  const int n0   = blockIdx.x * BN;
  const int lm   = lane & 15;
  const int kh   = lane >> 4;      // 0/1 lane half
  const int lrow = tid >> 2;       // 0..63
  const int lcol = (tid & 3) * 8;  // 0,8,16,24

  f8v acc[2][4] = {};

  auto loadTiles = [&](int kt, int buf) {
    const _Float16* ga = A  + (size_t)(m0 + lrow) * K + kt * BKK + lcol;
    const _Float16* gb = Bt + (size_t)(n0 + lrow) * K + kt * BKK + lcol;
    h8v a0 = *(const h8v*)ga;
    h8v a1 = *(const h8v*)(ga + (size_t)64 * K);
    h8v b0 = *(const h8v*)gb;
    h8v b1 = *(const h8v*)(gb + (size_t)64 * K);
    *(h8v*)&As[buf][lrow * LDT + lcol]        = a0;
    *(h8v*)&As[buf][(lrow + 64) * LDT + lcol] = a1;
    *(h8v*)&Bs[buf][lrow * LDT + lcol]        = b0;
    *(h8v*)&Bs[buf][(lrow + 64) * LDT + lcol] = b1;
  };

  loadTiles(0, 0);
  __syncthreads();
  const int nk = K / BKK;
  for (int kt = 0; kt < nk; ++kt) {
    const int cur = kt & 1;
    if (kt + 1 < nk) loadTiles(kt + 1, cur ^ 1);
    if (kt + 2 < nk) {  // global_prefetch_b8 of the tile after next
      __builtin_prefetch(A  + (size_t)(m0 + lrow) * K + (kt + 2) * BKK + lcol, 0, 0);
      __builtin_prefetch(Bt + (size_t)(n0 + lrow) * K + (kt + 2) * BKK + lcol, 0, 0);
    }
    const _Float16* ab = &As[cur][0];
    const _Float16* bb = &Bs[cur][0];
    h16v af[2];
#pragma unroll
    for (int mi = 0; mi < 2; ++mi) {
      const _Float16* p = ab + (wm * 32 + mi * 16 + lm) * LDT;
      af[mi] = mk16(*(const h8v*)(p + kh * 16), *(const h8v*)(p + kh * 16 + 8));
    }
    h16v bf[4];
#pragma unroll
    for (int ni = 0; ni < 4; ++ni) {
      const _Float16* p = bb + (wn * 64 + ni * 16 + lm) * LDT + kh * 16;
      bf[ni] = mk16(*(const h8v*)p, *(const h8v*)(p + 8));
    }
#pragma unroll
    for (int mi = 0; mi < 2; ++mi)
#pragma unroll
      for (int ni = 0; ni < 4; ++ni)
        acc[mi][ni] = wmma32(af[mi], bf[ni], acc[mi][ni]);
    __syncthreads();
  }

#pragma unroll
  for (int mi = 0; mi < 2; ++mi)
#pragma unroll
    for (int ni = 0; ni < 4; ++ni) {
      const int col = n0 + wn * 64 + ni * 16 + lm;
      const float bcol = bias[col];
#pragma unroll
      for (int r = 0; r < 8; ++r) {
        const int row = m0 + wm * 32 + mi * 16 + kh * 8 + r;
        float v = acc[mi][ni][r] + bcol;
        if (MODE == 1) v += res[(size_t)row * N + col];
        if (MODE == 3) v = gelu_tanh(v);
        if (MODE == 2 || MODE == 3) outH[(size_t)row * N + col] = (_Float16)v;
        else                        outF[(size_t)row * N + col] = v;
      }
    }
}

// ---------------------------------------------------------------------------
// Flash-style attention, one (batch,head,64-query-tile) per block.
// Scores phase: 8 waves as 4Mx2N over 64x64.  Output phase: 2Mx4N over 64xHD.
// Q/K tiles stream in via global_load_async_to_lds_b128 (ASYNCcnt); V goes
// through registers because it is transposed into LDS for the P*V B-operand.
// ---------------------------------------------------------------------------
template<int HD>
__global__ __launch_bounds__(256) void k_attn(
    const _Float16* __restrict__ Q, const _Float16* __restrict__ Kt,
    const _Float16* __restrict__ V, _Float16* __restrict__ O,
    int nh, float scale)
{
  constexpr int LQ = HD + 8;   // halves, rows stay 16B aligned
  constexpr int LV = 72;       // 64 + 8 halves
  constexpr int NT = HD / 64;  // per-wave output col tiles
  extern __shared__ __align__(16) char smem_raw[];
  _Float16* Qs = (_Float16*)smem_raw;      // 64 x LQ
  _Float16* Ks = Qs + 64 * LQ;             // 64 x LQ
  _Float16* Vt = Ks + 64 * LQ;             // HD x LV (V transposed)
  _Float16* Ph = Vt + HD * LV;             // 64 x LV probs f16
  float* Sf      = (float*)(Ph + 64 * LV); // 64 x 65 scores f32
  float* rowMax  = Sf + 64 * 65;
  float* rowSum  = rowMax + 64;
  float* rowCorr = rowSum + 64;

  const int tid  = threadIdx.x;
  const int lane = tid & 31, wave = tid >> 5;
  const int lm = lane & 15, kh = lane >> 4;
  const int swm = wave >> 1, swn = wave & 1;   // scores 4x2
  const int owm = wave >> 2, own = wave & 3;   // output 2x4
  const int b = blockIdx.x / nh, h = blockIdx.x % nh;
  const int m0 = blockIdx.y * 64;
  const size_t base = ((size_t)b * SEQ) * HTOT + (size_t)h * HD;

  // async Q tile load (memory -> LDS, no VGPR staging)
  for (int c = tid; c < 64 * HD / 8; c += 256) {
    const int r = c / (HD / 8);
    const int d = (c % (HD / 8)) * 8;
    async_copy_b128(&Qs[r * LQ + d], Q + base + (size_t)(m0 + r) * HTOT + d);
  }
  if (tid < 64) { rowMax[tid] = -3.0e38f; rowSum[tid] = 0.0f; }
  f8v o[2][NT] = {};
  wait_async0();
  __syncthreads();

  for (int kc = 0; kc < SEQ / 64; ++kc) {
    const int kb = kc * 64;
    // async K tile; V transposed through registers
    for (int c = tid; c < 64 * HD / 8; c += 256) {
      const int r = c / (HD / 8);
      const int d = (c % (HD / 8)) * 8;
      async_copy_b128(&Ks[r * LQ + d], Kt + base + (size_t)(kb + r) * HTOT + d);
      h8v vv = *(const h8v*)(V + base + (size_t)(kb + r) * HTOT + d);
#pragma unroll
      for (int j = 0; j < 8; ++j) Vt[(d + j) * LV + r] = vv[j];
    }
    wait_async0();
    __syncthreads();

    // S = scale * Q K^T  (16x32 per wave)
    f8v sacc[2] = {};
#pragma unroll
    for (int ks = 0; ks < HD / 32; ++ks) {
      const _Float16* ap = Qs + (swm * 16 + lm) * LQ + ks * 32;
      h16v aq = mk16(*(const h8v*)(ap + kh * 16), *(const h8v*)(ap + kh * 16 + 8));
#pragma unroll
      for (int ni = 0; ni < 2; ++ni) {
        const _Float16* bp = Ks + (swn * 32 + ni * 16 + lm) * LQ + ks * 32 + kh * 16;
        sacc[ni] = wmma32(aq, mk16(*(const h8v*)bp, *(const h8v*)(bp + 8)), sacc[ni]);
      }
    }
#pragma unroll
    for (int ni = 0; ni < 2; ++ni)
#pragma unroll
      for (int r = 0; r < 8; ++r)
        Sf[(swm * 16 + kh * 8 + r) * 65 + swn * 32 + ni * 16 + lm] =
            sacc[ni][r] * scale;
    __syncthreads();

    // online softmax, one row per thread 0..63
    if (tid < 64) {
      const float* srow = Sf + tid * 65;
      float mNew = rowMax[tid];
      for (int j = 0; j < 64; ++j) mNew = fmaxf(mNew, srow[j]);
      const float corr = __expf(rowMax[tid] - mNew);
      float s = 0.0f;
      _Float16* prow = Ph + tid * LV;
      for (int j = 0; j < 64; ++j) {
        float e = __expf(srow[j] - mNew);
        prow[j] = (_Float16)e;
        s += e;
      }
      rowSum[tid] = rowSum[tid] * corr + s;
      rowMax[tid] = mNew;
      rowCorr[tid] = corr;
    }
    __syncthreads();

    // rescale accumulators, then O += P * V
#pragma unroll
    for (int mi = 0; mi < 2; ++mi)
#pragma unroll
      for (int r = 0; r < 8; ++r) {
        const float corr = rowCorr[owm * 32 + mi * 16 + kh * 8 + r];
#pragma unroll
        for (int nj = 0; nj < NT; ++nj) o[mi][nj][r] *= corr;
      }
#pragma unroll
    for (int ks = 0; ks < 2; ++ks)
#pragma unroll
      for (int mi = 0; mi < 2; ++mi) {
        const _Float16* ap = Ph + (owm * 32 + mi * 16 + lm) * LV + ks * 32;
        h16v pa = mk16(*(const h8v*)(ap + kh * 16), *(const h8v*)(ap + kh * 16 + 8));
#pragma unroll
        for (int nj = 0; nj < NT; ++nj) {
          const _Float16* bp =
              Vt + (own * (HD / 4) + nj * 16 + lm) * LV + ks * 32 + kh * 16;
          o[mi][nj] = wmma32(pa, mk16(*(const h8v*)bp, *(const h8v*)(bp + 8)),
                             o[mi][nj]);
        }
      }
    __syncthreads();
  }

#pragma unroll
  for (int mi = 0; mi < 2; ++mi)
#pragma unroll
    for (int r = 0; r < 8; ++r) {
      const int rl = owm * 32 + mi * 16 + kh * 8 + r;
      const float inv = 1.0f / rowSum[rl];
#pragma unroll
      for (int nj = 0; nj < NT; ++nj) {
        const int col = own * (HD / 4) + nj * 16 + lm;
        O[base + (size_t)(m0 + rl) * HTOT + col] = (_Float16)(o[mi][nj][r] * inv);
      }
    }
}

// ---------------------------------------------------------------------------
// LayerNorm: one wave (32 lanes) per 512-wide row; f32 + f16 outputs.
// ---------------------------------------------------------------------------
__global__ __launch_bounds__(256) void k_ln(
    const float* __restrict__ in, const float* __restrict__ gamma,
    const float* __restrict__ beta, float* __restrict__ outF,
    _Float16* __restrict__ outH)
{
  const int lane = threadIdx.x & 31;
  const int row  = blockIdx.x * 8 + (threadIdx.x >> 5);
  const float* r = in + (size_t)row * HTOT;
  float v[16], s = 0.0f, s2 = 0.0f;
#pragma unroll
  for (int i = 0; i < 16; ++i) {
    v[i] = r[lane + i * 32];
    s += v[i]; s2 += v[i] * v[i];
  }
#pragma unroll
  for (int off = 16; off > 0; off >>= 1) {
    s  += __shfl_xor(s, off, 32);
    s2 += __shfl_xor(s2, off, 32);
  }
  const float mean = s * (1.0f / 512.0f);
  const float var  = s2 * (1.0f / 512.0f) - mean * mean;
  const float rstd = rsqrtf(var + 1e-6f);
#pragma unroll
  for (int i = 0; i < 16; ++i) {
    const int hc = lane + i * 32;
    const float ov = (v[i] - mean) * rstd * gamma[hc] + beta[hc];
    outF[(size_t)row * HTOT + hc] = ov;
    outH[(size_t)row * HTOT + hc] = (_Float16)ov;
  }
}

// f32 -> f16 elementwise
__global__ void k_cvt(const float* __restrict__ in, _Float16* __restrict__ out,
                      int n) {
  const int i = blockIdx.x * 256 + threadIdx.x;
  if (i < n) out[i] = (_Float16)in[i];
}

// W[K,N] f32 -> Wt[N,K] f16 (tiled transpose via LDS)
__global__ __launch_bounds__(256) void k_tcvt(const float* __restrict__ in,
                                              _Float16* __restrict__ out,
                                              int K, int N) {
  __shared__ _Float16 t[32][33];
  const int k0 = blockIdx.y * 32, n0 = blockIdx.x * 32;
  const int tx = threadIdx.x & 31, ty = threadIdx.x >> 5;
#pragma unroll
  for (int j = 0; j < 32; j += 8)
    t[ty + j][tx] = (_Float16)in[(size_t)(k0 + ty + j) * N + n0 + tx];
  __syncthreads();
#pragma unroll
  for (int j = 0; j < 32; j += 8)
    out[(size_t)(n0 + ty + j) * K + k0 + tx] = t[tx][ty + j];
}

// importance gate: mean-pool over sequence, sigmoid(pooled . w + b)
__global__ __launch_bounds__(256) void k_gate(
    const float* __restrict__ x1, const float* __restrict__ iw,
    const float* __restrict__ ib, float* __restrict__ impOut,
    float* __restrict__ stepsOut)
{
  __shared__ float red[256];
  const int b = blockIdx.x, tid = threadIdx.x;
  float acc = 0.0f;
  for (int hc = tid; hc < HTOT; hc += 256) {
    float s = 0.0f;
    const float* p = x1 + ((size_t)b * SEQ) * HTOT + hc;
    for (int si = 0; si < SEQ; ++si) s += p[(size_t)si * HTOT];
    acc += (s * (1.0f / 512.0f)) * iw[hc];
  }
  red[tid] = acc;
  __syncthreads();
  for (int st = 128; st > 0; st >>= 1) {
    if (tid < st) red[tid] += red[tid + st];
    __syncthreads();
  }
  if (tid == 0) {
    const float imp = 1.0f / (1.0f + __expf(-(red[0] + ib[0])));
    impOut[b]   = imp;
    stepsOut[b] = imp > 0.35f ? 6.0f : 0.0f;
  }
}

__global__ void k_select(const float* __restrict__ x1,
                         const float* __restrict__ hf,
                         const float* __restrict__ imp,
                         float* __restrict__ out) {
  const size_t i = (size_t)blockIdx.x * 256 + threadIdx.x;
  const int b = (int)(i / ((size_t)SEQ * HTOT));
  out[i] = (imp[b] > 0.35f) ? hf[i] : x1[i];
}

// ---------------------------------------------------------------------------
extern "C" void kernel_launch(void* const* d_in, const int* in_sizes, int n_in,
                              void* d_out, int out_size, void* d_ws,
                              size_t ws_size, hipStream_t stream)
{
  (void)in_sizes; (void)n_in; (void)out_size; (void)ws_size;
  const float* x      = (const float*)d_in[0];
  const float* s1_wq  = (const float*)d_in[1];
  const float* s1_bq  = (const float*)d_in[2];
  const float* s1_wk  = (const float*)d_in[3];
  const float* s1_bk  = (const float*)d_in[4];
  const float* s1_wv  = (const float*)d_in[5];
  const float* s1_bv  = (const float*)d_in[6];
  const float* s1_wo  = (const float*)d_in[7];
  const float* s1_bo  = (const float*)d_in[8];
  const float* s1_lns = (const float*)d_in[9];
  const float* s1_lnb = (const float*)d_in[10];
  const float* imp_w  = (const float*)d_in[11];
  const float* imp_b  = (const float*)d_in[12];
  const float* s2_wq  = (const float*)d_in[13];
  const float* s2_bq  = (const float*)d_in[14];
  const float* s2_wk  = (const float*)d_in[15];
  const float* s2_bk  = (const float*)d_in[16];
  const float* s2_wv  = (const float*)d_in[17];
  const float* s2_bv  = (const float*)d_in[18];
  const float* s2_wo  = (const float*)d_in[19];
  const float* s2_bo  = (const float*)d_in[20];
  const float* ln1_s  = (const float*)d_in[21];
  const float* ln1_b  = (const float*)d_in[22];
  const float* mlp_w1 = (const float*)d_in[23];
  const float* mlp_b1 = (const float*)d_in[24];
  const float* mlp_w2 = (const float*)d_in[25];
  const float* mlp_b2 = (const float*)d_in[26];
  const float* ln2_s  = (const float*)d_in[27];
  const float* ln2_b  = (const float*)d_in[28];

  float* out      = (float*)d_out;
  float* impOut   = out + (size_t)BATCH * SEQ * HTOT;
  float* stepsOut = impOut + BATCH;

  const size_t NA   = (size_t)BATCH * SEQ * HTOT;   // 4,194,304
  const size_t NHID = (size_t)BATCH * SEQ * 2048;

  // ---- workspace layout (f16 region first, then f32 region) ----
  _Float16* p16 = (_Float16*)d_ws;
  _Float16* wq1 = p16; p16 += 512 * 512;
  _Float16* wk1 = p16; p16 += 512 * 512;
  _Float16* wv1 = p16; p16 += 512 * 512;
  _Float16* wo1 = p16; p16 += 512 * 512;
  _Float16* wq2 = p16; p16 += 512 * 512;
  _Float16* wk2 = p16; p16 += 512 * 512;
  _Float16* wv2 = p16; p16 += 512 * 512;
  _Float16* wo2 = p16; p16 += 512 * 512;
  _Float16* w1t = p16; p16 += 512 * 2048;   // [2048][512]
  _Float16* w2t = p16; p16 += 2048 * 512;   // [512][2048]
  _Float16* C16a = p16; p16 += NA;          // x16 / h16 (cycles >= 1)
  _Float16* C16b = p16; p16 += NA;          // x1 f16
  _Float16* Q16  = p16; p16 += NA;
  _Float16* K16  = p16; p16 += NA;
  _Float16* V16  = p16; p16 += NA;
  _Float16* O16  = p16; p16 += NA;
  _Float16* L116 = p16; p16 += NA;          // h1 f16
  _Float16* HID16 = p16; p16 += NHID;       // gelu(mlp1) f16
  float* p32 = (float*)p16;
  float* X1F = p32; p32 += NA;
  float* HF  = p32; p32 += NA;
  float* H1F = p32; p32 += NA;
  float* T0F = p32; p32 += NA;

  // attention dynamic LDS sizes (>64KB allowed on CDNA5: 320KB/WGP)
  auto smemAttn = [](int hd) -> size_t {
    size_t halves = (size_t)64 * (hd + 8) * 2 + (size_t)hd * 72 + (size_t)64 * 72;
    return halves * 2 + (size_t)64 * 65 * 4 + 3 * 64 * 4;
  };
  const size_t sm64 = smemAttn(64), sm256 = smemAttn(256);
  (void)hipFuncSetAttribute(reinterpret_cast<const void*>(&k_attn<64>),
                            hipFuncAttributeMaxDynamicSharedMemorySize, (int)sm64);
  (void)hipFuncSetAttribute(reinterpret_cast<const void*>(&k_attn<256>),
                            hipFuncAttributeMaxDynamicSharedMemorySize, (int)sm256);

  // ---- weight prep: transpose + f16 convert ----
  auto TC = [&](const float* W, _Float16* Wt, int Kd, int Nd) {
    k_tcvt<<<dim3(Nd / 32, Kd / 32), 256, 0, stream>>>(W, Wt, Kd, Nd);
  };
  TC(s1_wq, wq1, 512, 512);  TC(s1_wk, wk1, 512, 512);
  TC(s1_wv, wv1, 512, 512);  TC(s1_wo, wo1, 512, 512);
  TC(s2_wq, wq2, 512, 512);  TC(s2_wk, wk2, 512, 512);
  TC(s2_wv, wv2, 512, 512);  TC(s2_wo, wo2, 512, 512);
  TC(mlp_w1, w1t, 512, 2048);
  TC(mlp_w2, w2t, 2048, 512);

  const dim3 g512(4, 64), g2048(16, 64);

  // ---- system 1: fast attention + residual + LN ----
  k_cvt<<<(int)(NA / 256), 256, 0, stream>>>(x, C16a, (int)NA);
  k_gemm<2><<<g512, 256, 0, stream>>>(C16a, wq1, s1_bq, nullptr, nullptr, Q16, 512, 512);
  k_gemm<2><<<g512, 256, 0, stream>>>(C16a, wk1, s1_bk, nullptr, nullptr, K16, 512, 512);
  k_gemm<2><<<g512, 256, 0, stream>>>(C16a, wv1, s1_bv, nullptr, nullptr, V16, 512, 512);
  k_attn<256><<<dim3(BATCH * 2, SEQ / 64), 256, sm256, stream>>>(
      Q16, K16, V16, O16, 2, 0.0625f);                       // 1/sqrt(256)
  k_gemm<1><<<g512, 256, 0, stream>>>(O16, wo1, s1_bo, x, T0F, nullptr, 512, 512);
  k_ln<<<NROWS / 8, 256, 0, stream>>>(T0F, s1_lns, s1_lnb, X1F, C16b);

  // ---- importance gate (also writes importance & steps outputs) ----
  k_gate<<<BATCH, 256, 0, stream>>>(X1F, imp_w, imp_b, impOut, stepsOut);

  // ---- system 2: 6 iterative transformer blocks ----
  for (int c = 0; c < 6; ++c) {
    const _Float16* cur16 = (c == 0) ? C16b : C16a;
    const float*    curF  = (c == 0) ? X1F : HF;
    k_gemm<2><<<g512, 256, 0, stream>>>(cur16, wq2, s2_bq, nullptr, nullptr, Q16, 512, 512);
    k_gemm<2><<<g512, 256, 0, stream>>>(cur16, wk2, s2_bk, nullptr, nullptr, K16, 512, 512);
    k_gemm<2><<<g512, 256, 0, stream>>>(cur16, wv2, s2_bv, nullptr, nullptr, V16, 512, 512);
    k_attn<64><<<dim3(BATCH * 8, SEQ / 64), 256, sm64, stream>>>(
        Q16, K16, V16, O16, 8, 0.125f);                      // 1/sqrt(64)
    k_gemm<1><<<g512, 256, 0, stream>>>(O16, wo2, s2_bo, curF, T0F, nullptr, 512, 512);
    k_ln<<<NROWS / 8, 256, 0, stream>>>(T0F, ln1_s, ln1_b, H1F, L116);
    k_gemm<3><<<g2048, 256, 0, stream>>>(L116, w1t, mlp_b1, nullptr, nullptr, HID16, 2048, 512);
    k_gemm<1><<<g512, 256, 0, stream>>>(HID16, w2t, mlp_b2, H1F, T0F, nullptr, 512, 2048);
    k_ln<<<NROWS / 8, 256, 0, stream>>>(T0F, ln2_s, ln2_b, HF, C16a);
  }

  // ---- final per-batch select ----
  k_select<<<(int)(NA / 256), 256, 0, stream>>>(X1F, HF, impOut, out);
}
